// HeimdallLocator_55954833932394
// MI455X (gfx1250) — matrix operand order repaired and checked by reading, compiled, not run on hardware
//
#include <hip/hip_runtime.h>
#include <hip/hip_bf16.h>

#define HIDDEN     256
#define RANK       128
#define N_NODES    12800
#define T_DIM      32
#define NUM_GRAPHS 256
#define TOTAL_PIX  32768
#define IMG0       16384
#define IMG01      24576
#define LN_EPS     1e-5f

// LDS row pitch for staged A: 128 halfs + 8 pad -> 272B; 272/4=68 dwords == 4 mod 64
// so the 16 rows hit disjoint 4-bank groups under ds_load_b128 (conflict-minimal).
#define A_PITCH    136

typedef __attribute__((ext_vector_type(16))) _Float16 v16h;
typedef __attribute__((ext_vector_type(8)))  _Float16 v8h;
typedef __attribute__((ext_vector_type(8)))  float    v8f;
typedef __attribute__((ext_vector_type(4)))  float    v4f;   // native vector for NT builtins

__device__ __forceinline__ float fast_sigmoid(float v)
{
    // v_exp_f32 + v_rcp_f32 instead of the IEEE div sequence (12+ VALU ops)
    return __builtin_amdgcn_rcpf(1.0f + __expf(-v));
}

// ---------------------------------------------------------------------------
// K1: h = mean_T(x); alpha = h . score_w + score_b   (one block per node)
// x is 419MB read exactly once -> TH=NT loads so it never displaces L2 data.
// ---------------------------------------------------------------------------
__global__ void k_reduce_score(const float* __restrict__ x,
                               const float* __restrict__ score_w,
                               const float* __restrict__ score_b,
                               float* __restrict__ h,
                               float* __restrict__ alpha)
{
    const int node = blockIdx.x;
    const int f = threadIdx.x;                      // 0..255
    const v4f* xp = (const v4f*)(x + ((size_t)node * HIDDEN + f) * T_DIM);
    float s = 0.f;
#pragma unroll
    for (int i = 0; i < T_DIM / 4; ++i) {
        v4f v = __builtin_nontemporal_load(&xp[i]);
        s += (v.x + v.y) + (v.z + v.w);
    }
    const float hm = s * (1.0f / T_DIM);
    h[(size_t)node * HIDDEN + f] = hm;              // RT store: stays L2-resident for K2

    // alpha = dot(h_row, score_w): wave32 butterfly, then 8 partials via LDS
    float v = hm * score_w[f];
#pragma unroll
    for (int off = 16; off > 0; off >>= 1) v += __shfl_xor(v, off, 32);
    __shared__ float wred[8];
    if ((f & 31) == 0) wred[f >> 5] = v;
    __syncthreads();
    if (f == 0) {
        float t = wred[0];
#pragma unroll
        for (int w = 1; w < 8; ++w) t += wred[w];
        alpha[node] = t + score_b[0];
    }
}

// ---------------------------------------------------------------------------
// K2: per-graph segment softmax + weighted pooling (one block per graph).
// ---------------------------------------------------------------------------
__device__ __forceinline__ int lower_bound_dev(const int* __restrict__ b, int key)
{
    int lo = 0, hi = N_NODES;
    while (lo < hi) { int mid = (lo + hi) >> 1; if (b[mid] < key) lo = mid + 1; else hi = mid; }
    return lo;
}

__global__ void k_segpool(const float* __restrict__ h,
                          const float* __restrict__ alpha,
                          const int* __restrict__ batch,
                          float* __restrict__ pooled)
{
    const int g = blockIdx.x, tid = threadIdx.x;
    __shared__ int s_se[2];
    if (tid == 0) s_se[0] = lower_bound_dev(batch, g);
    if (tid == 1) s_se[1] = lower_bound_dev(batch, g + 1);
    __syncthreads();
    const int start = s_se[0], end = s_se[1];

    __shared__ float red[HIDDEN];
    float m = -3.4e38f;
    for (int i = start + tid; i < end; i += HIDDEN) m = fmaxf(m, alpha[i]);
    red[tid] = m; __syncthreads();
    for (int off = HIDDEN / 2; off > 0; off >>= 1) {
        if (tid < off) red[tid] = fmaxf(red[tid], red[tid + off]);
        __syncthreads();
    }
    const float mx = red[0];
    __syncthreads();

    float s = 0.f;
    for (int i = start + tid; i < end; i += HIDDEN) s += __expf(alpha[i] - mx);
    red[tid] = s; __syncthreads();
    for (int off = HIDDEN / 2; off > 0; off >>= 1) {
        if (tid < off) red[tid] += red[tid + off];
        __syncthreads();
    }
    const float denom = red[0];
    __syncthreads();
    const float inv = denom > 0.f ? 1.0f / denom : 0.0f;

    __shared__ float wbuf[HIDDEN];
    float acc = 0.f;                                // thread tid owns feature tid
    for (int base = start; base < end; base += HIDDEN) {
        const int n = min(HIDDEN, end - base);
        wbuf[tid] = (tid < n) ? __expf(alpha[base + tid] - mx) * inv : 0.0f;
        __syncthreads();
        for (int j = 0; j < n; ++j)
            acc += h[(size_t)(base + j) * HIDDEN + tid] * wbuf[j];
        __syncthreads();
    }
    pooled[g * HIDDEN + tid] = acc;
}

// ---------------------------------------------------------------------------
// K3: fused MLP heads (one block per graph): loc_w1+LN+relu, loc_w2+relu->f16,
//     coord path + sigmoid coords.
// ---------------------------------------------------------------------------
__global__ void k_heads(const float* __restrict__ pooled,
                        const float* __restrict__ loc_w1, const float* __restrict__ loc_b1,
                        const float* __restrict__ ln_g,  const float* __restrict__ ln_b,
                        const float* __restrict__ loc_w2,
                        const float* __restrict__ coord_w1, const float* __restrict__ coord_b1,
                        const float* __restrict__ coord_w2, const float* __restrict__ coord_b2,
                        _Float16* __restrict__ z2h, float* __restrict__ coords)
{
    const int g = blockIdx.x, tid = threadIdx.x;
    __shared__ float p[HIDDEN];
    __shared__ float buf[HIDDEN];
    __shared__ float red[HIDDEN];

    p[tid] = pooled[g * HIDDEN + tid];
    __syncthreads();

    float z = loc_b1[tid];
    {
        const float4* wv = (const float4*)(loc_w1 + (size_t)tid * HIDDEN);
        const float4* pv = (const float4*)p;
#pragma unroll 8
        for (int i = 0; i < HIDDEN / 4; ++i) {
            float4 a = wv[i], b = pv[i];
            z += a.x * b.x + a.y * b.y + a.z * b.z + a.w * b.w;
        }
    }
    red[tid] = z; __syncthreads();
    for (int off = HIDDEN / 2; off > 0; off >>= 1) {
        if (tid < off) red[tid] += red[tid + off];
        __syncthreads();
    }
    const float mu = red[0] * (1.0f / HIDDEN);
    __syncthreads();
    red[tid] = z * z; __syncthreads();
    for (int off = HIDDEN / 2; off > 0; off >>= 1) {
        if (tid < off) red[tid] += red[tid + off];
        __syncthreads();
    }
    const float var = red[0] * (1.0f / HIDDEN) - mu * mu;
    __syncthreads();

    const float zn = (z - mu) * rsqrtf(var + LN_EPS) * ln_g[tid] + ln_b[tid];
    buf[tid] = fmaxf(zn, 0.0f);
    __syncthreads();

    if (tid < RANK) {
        float s = 0.f;
        const float4* wv = (const float4*)(loc_w2 + (size_t)tid * HIDDEN);
        const float4* bv = (const float4*)buf;
#pragma unroll 8
        for (int i = 0; i < HIDDEN / 4; ++i) {
            float4 a = wv[i], b = bv[i];
            s += a.x * b.x + a.y * b.y + a.z * b.z + a.w * b.w;
        }
        z2h[g * RANK + tid] = (_Float16)fmaxf(s, 0.0f);
    }

    float c = coord_b1[tid];
    {
        const float4* wv = (const float4*)(coord_w1 + (size_t)tid * HIDDEN);
        const float4* pv = (const float4*)p;
#pragma unroll 8
        for (int i = 0; i < HIDDEN / 4; ++i) {
            float4 a = wv[i], b = pv[i];
            c += a.x * b.x + a.y * b.y + a.z * b.z + a.w * b.w;
        }
    }
    __syncthreads();                  // z2 readers done with buf
    buf[tid] = fmaxf(c, 0.0f);
    __syncthreads();
    if (tid < 3) {
        float s = coord_b2[tid];
        const float* wv = coord_w2 + tid * HIDDEN;
        for (int i = 0; i < HIDDEN; ++i) s += wv[i] * buf[i];
        coords[g * 3 + tid] = fast_sigmoid(s);
    }
}

// ---------------------------------------------------------------------------
// K4: loc_w3 f32 -> f16 (one pass; NT reads, RT writes so w3h parks in L2)
// ---------------------------------------------------------------------------
__global__ void k_w3_to_half(const float* __restrict__ w, _Float16* __restrict__ wh)
{
    const int i = (blockIdx.x * blockDim.x + threadIdx.x) * 4;
    v4f v = __builtin_nontemporal_load((const v4f*)(w + i));
    wh[i + 0] = (_Float16)v.x;
    wh[i + 1] = (_Float16)v.y;
    wh[i + 2] = (_Float16)v.z;
    wh[i + 3] = (_Float16)v.w;
}

// ---------------------------------------------------------------------------
// K5: out(256,32768) = sigmoid(z2(256,128) @ w3^T + b3), fused scatter into
//     img0/img1/img2 regions of d_out.
// One wave32 per 16-column N-tile: B fragments (4x v16h) live in registers for
// the whole kernel; A (256x128 f16) staged into LDS in two 34KB phases and
// reused by all 8 waves; 64 v_wmma_f32_16x16x32_f16 per wave. Output streamed
// with nontemporal stores (33MB, never re-read).
// ---------------------------------------------------------------------------
__global__ void k_wmma_out(const _Float16* __restrict__ A,   // (256,128)   z2 f16
                           const _Float16* __restrict__ B,   // (32768,128) w3 f16
                           const float* __restrict__ bias,   // (32768)
                           float* __restrict__ out)
{
    const int lane = threadIdx.x & 31;
    const int wave = threadIdx.x >> 5;
    const int nt = blockIdx.x * 8 + wave;   // 0..2047 (N tiles)
    const int m   = lane & 15;
    const int khi = lane >> 4;

    // --- B fragments: lane khi*16+n holds K = j + 16*khi (contiguous order, same as A), K=0..127
    const _Float16* bp = B + (size_t)(nt * 16 + m) * RANK + 16 * khi;
    v16h b0 = *(const v16h*)(bp);
    v16h b1 = *(const v16h*)(bp + 32);
    v16h b2 = *(const v16h*)(bp + 64);
    v16h b3 = *(const v16h*)(bp + 96);

    // --- epilogue constants for this wave's 16 output columns
    const int col = nt * 16 + m;
    const float bv = bias[col];
    size_t seg; int ccol, cstr;
    if (col < IMG0)       { seg = 0;                            ccol = col;         cstr = IMG0; }
    else if (col < IMG01) { seg = (size_t)NUM_GRAPHS * IMG0;    ccol = col - IMG0;  cstr = 8192; }
    else                  { seg = (size_t)NUM_GRAPHS * IMG01;   ccol = col - IMG01; cstr = 8192; }
    // this lane's column base; rows advance by cstr floats
    float* const colbase = out + seg + ccol + (size_t)khi * 8 * cstr;

    __shared__ __align__(16) _Float16 sA[128 * A_PITCH];   // 34,816 B

    for (int phase = 0; phase < 2; ++phase) {
        // cooperative copy of A rows [phase*128, phase*128+128): 1024 uint4
        const uint4* Ag = (const uint4*)(A + (size_t)phase * 128 * RANK);
#pragma unroll
        for (int k = 0; k < 4; ++k) {
            const int i = threadIdx.x + k * 256;  // 0..1023
            const int row = i >> 3, s16 = i & 7;  // 8x 16B segments per 128B row
            *(uint4*)((char*)sA + row * (A_PITCH * 2) + s16 * 16) = Ag[i];
        }
        __syncthreads();

#pragma unroll 4
        for (int mtl = 0; mtl < 8; ++mtl) {
            // A 16x32 layout: lane khi*16+m holds K = j + 16*khi (matches B)
            const _Float16* ap = sA + (mtl * 16 + m) * A_PITCH + 16 * khi;
            v8f acc = {};
#pragma unroll
            for (int kt = 0; kt < RANK; kt += 32) {
                v8h a0 = *(const v8h*)(ap + kt);        // ds_load_b128
                v8h a1 = *(const v8h*)(ap + kt + 8);
                v16h a;
#pragma unroll
                for (int j = 0; j < 8; ++j) { a[j] = a0[j]; a[8 + j] = a1[j]; }
                const v16h bfrag = (kt == 0) ? b0 : (kt == 32) ? b1 : (kt == 64) ? b2 : b3;
                acc = __builtin_amdgcn_wmma_f32_16x16x32_f16(
                    false, a, false, bfrag, (short)0, acc, false, false);
            }
            // acc[r] -> out[g = phase*128 + mtl*16 + 8*khi + r][col]; rows stride cstr
            float* rp = colbase + (size_t)(phase * 128 + mtl * 16) * cstr;
#pragma unroll
            for (int r = 0; r < 8; ++r) {
                __builtin_nontemporal_store(fast_sigmoid(acc[r] + bv), rp);
                rp += cstr;
            }
        }
        __syncthreads();
    }
}

// ---------------------------------------------------------------------------
extern "C" void kernel_launch(void* const* d_in, const int* in_sizes, int n_in,
                              void* d_out, int out_size, void* d_ws, size_t ws_size,
                              hipStream_t stream)
{
    const float* x        = (const float*)d_in[0];
    const int*   batch    = (const int*)  d_in[1];
    const float* score_w  = (const float*)d_in[2];
    const float* score_b  = (const float*)d_in[3];
    const float* loc_w1   = (const float*)d_in[4];
    const float* loc_b1   = (const float*)d_in[5];
    const float* ln_g     = (const float*)d_in[6];
    const float* ln_b     = (const float*)d_in[7];
    const float* loc_w2   = (const float*)d_in[8];
    const float* loc_w3   = (const float*)d_in[9];
    const float* loc_b3   = (const float*)d_in[10];
    const float* coord_w1 = (const float*)d_in[11];
    const float* coord_b1 = (const float*)d_in[12];
    const float* coord_w2 = (const float*)d_in[13];
    const float* coord_b2 = (const float*)d_in[14];

    char* ws = (char*)d_ws;                       // all offsets 256B-aligned
    float*    h      = (float*)   (ws);                  // 12800*256*4  = 13107200
    float*    alpha  = (float*)   (ws + 13107200);       // 12800*4      = 51200
    float*    pooled = (float*)   (ws + 13158400);       // 256*256*4    = 262144
    _Float16* z2h    = (_Float16*)(ws + 13420544);       // 256*128*2    = 65536
    _Float16* w3h    = (_Float16*)(ws + 13486080);       // 32768*128*2  = 8388608

    float* out    = (float*)d_out;
    float* coords = out + (size_t)NUM_GRAPHS * TOTAL_PIX;

    k_reduce_score<<<N_NODES, HIDDEN, 0, stream>>>(x, score_w, score_b, h, alpha);
    k_w3_to_half<<<(TOTAL_PIX * RANK) / (256 * 4), 256, 0, stream>>>(loc_w3, w3h);
    k_segpool<<<NUM_GRAPHS, HIDDEN, 0, stream>>>(h, alpha, batch, pooled);
    k_heads<<<NUM_GRAPHS, HIDDEN, 0, stream>>>(pooled, loc_w1, loc_b1, ln_g, ln_b, loc_w2,
                                               coord_w1, coord_b1, coord_w2, coord_b2,
                                               z2h, coords);
    k_wmma_out<<<TOTAL_PIX / 16 / 8, 256, 0, stream>>>(z2h, w3h, loc_b3, out);
}
